// LSTM_66726611911224
// MI455X (gfx1250) — matrix operand order repaired and checked by reading, not compiled, not run on hardware
//
#include <hip/hip_runtime.h>

// ---------------------------------------------------------------------------
// LSTM on gfx1250 (MI455X): persistent per-batch-tile scan with bf16 WMMA.
//   grid = 4 workgroups x 256 threads (8 wave32); WG g owns batch rows
//   [16g, 16g+16). Fused: gates GEMM + cell update + output GEMM per step.
//   Round 3: prefetch with locality 3 (WGP-scope) instead of SYS-scope.
// ---------------------------------------------------------------------------

typedef __attribute__((ext_vector_type(16))) __bf16        v16bf;
typedef __attribute__((ext_vector_type(8)))  float         v8f;
typedef __attribute__((ext_vector_type(4)))  unsigned int  v4u;
typedef __attribute__((ext_vector_type(4)))  float         v4f;

#define T_STEPS 2048
#define BATCH   64
#define IN_F    128
#define HID     256
#define OUT_F   128
#define KTOT    384            // IN_F + HID (fused gate-GEMM K)
#define NKT     12             // KTOT / 32
#define NNT     64             // 4H / 16 N-tiles
#define WBLK_DW 256            // dwords per packed 16x32 bf16 B-fragment block
#define WBUF_DW (NNT * NKT * WBLK_DW)   // 196608 dwords (768 KB)
#define WOBUF_DW (8 * 8 * WBLK_DW)      // 16384 dwords (64 KB)
#define OUT_ELEMS ((size_t)T_STEPS * BATCH * OUT_F)  // 16777216

__device__ __forceinline__ unsigned f2bf(float f) {
  // float -> bf16 bits, round-to-nearest-even
  unsigned u = __float_as_uint(f);
  return (u + 0x7FFFu + ((u >> 16) & 1u)) >> 16;
}

__device__ __forceinline__ float sigm(float x) {
  return 1.0f / (1.0f + __expf(-x));
}

__device__ __forceinline__ float tanh_f(float x) {
  x = fminf(fmaxf(x, -15.0f), 15.0f);
  float e = __expf(2.0f * x);
  return (e - 1.0f) / (e + 1.0f);
}

union FragU { v16bf bf; v4u u[2]; };

// ---------------------------------------------------------------------------
// Kernel 0: pre-swizzle [W_ih | W_hh] and W_out into WMMA B-fragment layout.
// Block (nt,kt) = 16x32 bf16 tile; within block: lane l holds N = nt*16+l%16,
// K = kt*32 + (l/16)*16 + 2v (+1 in high half of dword v). 1 KB per block.
// ---------------------------------------------------------------------------
__global__ void lstm_pack_weights(const float* __restrict__ W_ih,
                                  const float* __restrict__ W_hh,
                                  const float* __restrict__ W_out,
                                  unsigned* __restrict__ wbuf,
                                  unsigned* __restrict__ wobuf) {
  int id = blockIdx.x * 256 + threadIdx.x;
  if (id < NNT * NKT * 256) {
    int block = id >> 8, rem = id & 255;
    int lane = rem >> 3, v = rem & 7;
    int nt = block / NKT, kt = block - nt * NKT;
    int g = nt * 16 + (lane & 15);                    // gate row (0..1023)
    int K = kt * 32 + ((lane >> 4) << 4) + 2 * v;     // fused-K (0..382, even)
    float w0, w1;
    if (K < IN_F) { w0 = W_ih[g * IN_F + K];          w1 = W_ih[g * IN_F + K + 1]; }
    else          { w0 = W_hh[g * HID + (K - IN_F)];  w1 = W_hh[g * HID + (K - IN_F) + 1]; }
    wbuf[(block << 8) + (lane << 3) + v] = f2bf(w0) | (f2bf(w1) << 16);
  } else {
    id -= NNT * NKT * 256;
    if (id < 8 * 8 * 256) {
      int block = id >> 8, rem = id & 255;
      int lane = rem >> 3, v = rem & 7;
      int nt = block >> 3, kt = block & 7;
      int o = nt * 16 + (lane & 15);                  // out col (0..127)
      int j = kt * 32 + ((lane >> 4) << 4) + 2 * v;   // hidden idx (0..254, even)
      float w0 = W_out[o * HID + j], w1 = W_out[o * HID + j + 1];
      wobuf[(block << 8) + (lane << 3) + v] = f2bf(w0) | (f2bf(w1) << 16);
    }
  }
}

// ---------------------------------------------------------------------------
// Kernel 1: persistent LSTM scan. One WG per 16 batch rows.
// ---------------------------------------------------------------------------
__global__ __launch_bounds__(256, 2)
void lstm_scan(const float* __restrict__ x,
               const float* __restrict__ b_ih, const float* __restrict__ b_hh,
               const float* __restrict__ b_out,
               const unsigned* __restrict__ wbuf,
               const unsigned* __restrict__ wobuf,
               float* __restrict__ out) {
  // A-operand staging: [16 rows][384 K] bf16, row-major (12 KB)
  __shared__ __align__(16) unsigned short Abuf[16 * KTOT];

  const int tid  = threadIdx.x;
  const int wg   = blockIdx.x;        // batch tile: rows wg*16 ..
  const int wave = tid >> 5;          // 0..7
  const int lane = tid & 31;
  const int nl   = lane & 15;
  const int hs   = lane >> 4;         // K/M half-select per WMMA layout

  // Per-lane gate biases: gate G, sub-tile tl -> column G*256 + wave*32 + tl*16 + nl
  float bsum[8];
  #pragma unroll
  for (int G = 0; G < 4; ++G)
    #pragma unroll
    for (int tl = 0; tl < 2; ++tl) {
      int idx = G * HID + wave * 32 + tl * 16 + nl;
      bsum[G * 2 + tl] = b_ih[idx] + b_hh[idx];
    }
  float bo = b_out[wave * 16 + nl];

  // h0 = 0: zero the h region of Abuf (cols 128..383)
  for (int i = tid; i < 16 * 128; i += 256) {
    int r = i >> 7, c = i & 127;
    ((unsigned*)Abuf)[r * (KTOT / 2) + (IN_F / 2) + c] = 0u;
  }
  // c0 = 0, kept in registers for the entire scan
  float cst[16];
  #pragma unroll
  for (int i = 0; i < 16; ++i) cst[i] = 0.0f;

  // stage x_t (fp32 -> bf16) into Abuf cols 0..127; 8 elems per thread
  auto load_x = [&](int t) {
    int r = tid >> 4, k0 = (tid & 15) << 3;
    const float* px = x + (((size_t)t * BATCH + wg * 16 + r) * IN_F + k0);
    v4f a = *(const v4f*)px;
    v4f b = *(const v4f*)(px + 4);
    v4u d;
    d.x = f2bf(a.x) | (f2bf(a.y) << 16);
    d.y = f2bf(a.z) | (f2bf(a.w) << 16);
    d.z = f2bf(b.x) | (f2bf(b.y) << 16);
    d.w = f2bf(b.z) | (f2bf(b.w) << 16);
    *(v4u*)(Abuf + r * KTOT + k0) = d;
  };

  load_x(0);
  __syncthreads();

  const v8f vzero = {0.f, 0.f, 0.f, 0.f, 0.f, 0.f, 0.f, 0.f};

  for (int t = 0; t < T_STEPS; ++t) {
    // ---- gates GEMM: [16 x 384] x [384 x 1024], wave owns 8 N-tiles ----
    v8f acc[8];
    #pragma unroll
    for (int n = 0; n < 8; ++n) acc[n] = vzero;

    // unroll capped at 2: keeps B-fragments-in-flight (and VGPR pressure) low
    #pragma unroll 2
    for (int kt = 0; kt < NKT; ++kt) {
      FragU a;
      int ab = nl * KTOT + kt * 32 + hs * 16;
      a.u[0] = *(const v4u*)(Abuf + ab);        // K = kt*32 + hs*16 .. +7
      a.u[1] = *(const v4u*)(Abuf + ab + 8);    // K = +8 .. +15
      if (kt + 1 < NKT)  // locality 3 -> WGP-scope RT prefetch of next K-slab
        __builtin_prefetch(wbuf + ((((2 * wave) * NKT + kt + 1) << 8) + (lane << 3)), 0, 3);
      #pragma unroll
      for (int G = 0; G < 4; ++G) {
        #pragma unroll
        for (int tl = 0; tl < 2; ++tl) {
          int nt = G * 16 + 2 * wave + tl;      // wave owns j in [32w, 32w+32) of every gate
          FragU b;
          const v4u* p = (const v4u*)(wbuf + ((nt * NKT + kt) << 8) + (lane << 3));
          b.u[0] = p[0];
          b.u[1] = p[1];
          acc[G * 2 + tl] = __builtin_amdgcn_wmma_f32_16x16x32_bf16(
              false, a.bf, false, b.bf, (short)0, acc[G * 2 + tl], false, false);
        }
      }
    }
    __syncthreads();  // all A-buffer reads (gate + prev y GEMM) complete

    // ---- cell update: i/f/g/o for the same hidden unit are in the same
    //      lane/element of four C-frags -> pure register math ----
    #pragma unroll
    for (int tl = 0; tl < 2; ++tl) {
      #pragma unroll
      for (int v = 0; v < 8; ++v) {
        float gi = acc[0 + tl][v] + bsum[0 + tl];
        float gf = acc[2 + tl][v] + bsum[2 + tl];
        float gg = acc[4 + tl][v] + bsum[4 + tl];
        float go = acc[6 + tl][v] + bsum[6 + tl];
        float i_ = sigm(gi), f_ = sigm(gf), g_ = tanh_f(gg), o_ = sigm(go);
        float c_ = f_ * cst[tl * 8 + v] + i_ * g_;
        cst[tl * 8 + v] = c_;
        float h_ = o_ * tanh_f(c_);
        int M = v + (hs << 3);                       // batch row within tile
        int j = wave * 32 + tl * 16 + nl;            // hidden unit
        Abuf[M * KTOT + IN_F + j] = (unsigned short)f2bf(h_);
        if (t == T_STEPS - 1) {
          int brow = wg * 16 + M;
          out[OUT_ELEMS + brow * HID + j] = h_;                    // h_T
          out[OUT_ELEMS + BATCH * HID + brow * HID + j] = c_;      // c_T
        }
      }
    }
    if (t + 1 < T_STEPS) load_x(t + 1);   // cols 0..127, disjoint from y-GEMM reads
    __syncthreads();  // h_t visible in Abuf

    // ---- output GEMM: y = h_t @ W_out^T + b_out.
    //      B-fragments streamed from L2-resident packed W_out. ----
    v8f yacc = vzero;
    #pragma unroll 2
    for (int kt = 0; kt < 8; ++kt) {
      FragU a;
      int ab = nl * KTOT + (kt + 4) * 32 + hs * 16;  // K-tiles 4..11 == h region
      a.u[0] = *(const v4u*)(Abuf + ab);
      a.u[1] = *(const v4u*)(Abuf + ab + 8);
      FragU b;
      const v4u* p = (const v4u*)(wobuf + (((wave << 3) + kt) << 8) + (lane << 3));
      b.u[0] = p[0];
      b.u[1] = p[1];
      if (kt + 1 < 8)
        __builtin_prefetch(wobuf + ((((wave << 3) + kt + 1) << 8) + (lane << 3)), 0, 3);
      yacc = __builtin_amdgcn_wmma_f32_16x16x32_bf16(
          false, a.bf, false, b.bf, (short)0, yacc, false, false);
    }
    float* po = out + ((size_t)t * BATCH + wg * 16) * OUT_F + wave * 16 + nl;
    #pragma unroll
    for (int v = 0; v < 8; ++v) {
      int M = v + (hs << 3);
      po[(size_t)M * OUT_F] = yacc[v] + bo;
    }
  }
}

// ---------------------------------------------------------------------------
extern "C" void kernel_launch(void* const* d_in, const int* in_sizes, int n_in,
                              void* d_out, int out_size, void* d_ws, size_t ws_size,
                              hipStream_t stream) {
  const float* input = (const float*)d_in[0];   // [T, B, IN]
  const float* W_ih  = (const float*)d_in[1];   // [4H, IN]
  const float* W_hh  = (const float*)d_in[2];   // [4H, H]
  const float* b_ih  = (const float*)d_in[3];   // [4H]
  const float* b_hh  = (const float*)d_in[4];   // [4H]
  const float* W_out = (const float*)d_in[5];   // [OUT, H]
  const float* b_out = (const float*)d_in[6];   // [OUT]
  float* out = (float*)d_out;

  unsigned* wbuf  = (unsigned*)d_ws;            // 768 KB packed gate weights
  unsigned* wobuf = wbuf + WBUF_DW;             // 64 KB packed W_out

  int pack_threads = NNT * NKT * 256 + 8 * 8 * 256;   // 212992
  lstm_pack_weights<<<(pack_threads + 255) / 256, 256, 0, stream>>>(
      W_ih, W_hh, W_out, wbuf, wobuf);

  lstm_scan<<<4, 256, 0, stream>>>(input, b_ih, b_hh, b_out, wbuf, wobuf, out);
}
